// AmpereFP8Linear_81801947119896
// MI455X (gfx1250) — matrix operand order repaired and checked by reading, not compiled, not run on hardware
//
#include <hip/hip_runtime.h>

// Fused int8-dequant + GEMM for MI455X (gfx1250, wave32, WMMA + TDM).
// out[m,n] = (sum_k act_f16[m,k] * (f16)w_i8[k,n]) * scale[n], fp32 accum, f16 out.
//
// Compute-bound (0.48 TFLOP vs ~9us HBM; A+W fit in 192MB L2). Structure:
//  - 16x v_wmma_f32_16x16x32_f16 per K-step (64x64 per-wave tile, 4x2 wave grid)
//  - A tile moved by the Tensor Data Mover (tensor_load_to_lds, TENSORcnt):
//    one instruction DMAs the whole 256x32 f16 tile to LDS with native
//    pad_interval/pad_amount generating the 80B bank-skewed row pitch.
//  - B tile software-pipelined through VGPRs (int8->f16 cvt co-executes with XDL)
//  - double-buffered LDS, one barrier per K-step.

typedef __attribute__((ext_vector_type(16))) _Float16    v16h;
typedef __attribute__((ext_vector_type(8)))  _Float16    v8h;
typedef __attribute__((ext_vector_type(8)))  float       v8f;
typedef __attribute__((ext_vector_type(16))) signed char c16;
typedef __attribute__((ext_vector_type(4)))  unsigned int v4u;
typedef __attribute__((ext_vector_type(8)))  int          v8i;
typedef __attribute__((ext_vector_type(4)))  int          v4i;

constexpr int MDIM = 4096;
constexpr int KDIM = 4096;
constexpr int NDIM = 14336;

constexpr int BM = 256;       // large BM: dequant VALU per WMMA flop ~ 1/BM
constexpr int BN = 128;
constexpr int BK = 32;        // == WMMA K for f16
constexpr int AKP = BK + 8;   // A LDS pitch (halves): 64B row + 16B pad (TDM pad fields)
constexpr int BKP = BK + 8;   // B LDS pitch (halves): [n][k] transposed layout

__global__ __launch_bounds__(256)
void fused_dq_gemm_f16(const _Float16* __restrict__ A,      // [M,K] fp16
                       const signed char* __restrict__ W,   // [K,N] int8
                       const _Float16* __restrict__ S,      // [N]   fp16
                       _Float16* __restrict__ O)            // [M,N] fp16
{
    __shared__ _Float16 As[2][BM * AKP];   // 2 x 20480 B
    __shared__ _Float16 Bs[2][BN * BKP];   // 2 x 10240 B  (61440 B of 320KB)

    const int tid  = threadIdx.x;
    const int lane = tid & 31;
    const int wave = tid >> 5;
    const int wm   = wave >> 1;          // 0..3 -> 64 M-rows each
    const int wn   = wave & 1;           // 0..1 -> 64 N-cols each

    const long bm = (long)blockIdx.y * BM;
    const long bn = (long)blockIdx.x * BN;

    const int row16 = lane & 15;
    const int kgrp  = lane >> 4;

    const v8f vzero = {};
    v8f acc[4][4];
#pragma unroll
    for (int mt = 0; mt < 4; ++mt)
#pragma unroll
        for (int nt = 0; nt < 4; ++nt)
            acc[mt][nt] = vzero;

    // B staging: one 16B int8 chunk/thread -> 32 k-rows x 128 n-cols
    const int b_k = tid >> 3;
    const int b_n = (tid & 7) * 16;

    // ---- TDM: DMA one 256x32 f16 tile (global, row-major, stride K) into LDS
    // with 16B padding after each 64B row -> 80B pitch (= AKP halves).
    const unsigned as_base[2] = {
        (unsigned)(size_t)(void*)&As[0][0],
        (unsigned)(size_t)(void*)&As[1][0],
    };
    auto issue_tdm_a = [&](int bufIdx, int k0) {
        if (wave == 0) {
            const unsigned long long ga =
                (unsigned long long)(const void*)(A + bm * (long)KDIM + k0);
            // D# group0: count=1 | lds_addr | global_addr(56:0) | type=2
            const v4u g0 = { 1u, as_base[bufIdx],
                             (unsigned)ga, (unsigned)(ga >> 32) | 0x80000000u };
            // D# group1: data_size=1(2B), pad_enable, pad_interval=3 (16 DW = 64B),
            // pad_amount=3 (4 DW = 16B); tensor_dim0=K, tensor_dim1=M;
            // tile_dim0=32, tile_dim1=256; tensor_dim0_stride=K.
            const v8i g1 = { (int)0x06D10000,
                             (int)((KDIM & 0xFFFF) << 16),
                             (int)(((KDIM >> 16) & 0xFFFF) | ((MDIM & 0xFFFF) << 16)),
                             (int)(((MDIM >> 16) & 0xFFFF) | (BK << 16)),
                             (int)BM,
                             (int)KDIM,
                             0, 0 };
            const v4i z4 = {};
#if defined(__clang_major__) && (__clang_major__ >= 23)
            const v8i z8 = {};
            __builtin_amdgcn_tensor_load_to_lds(g0, g1, z4, z4, z8, 0);
#else
            __builtin_amdgcn_tensor_load_to_lds(g0, g1, z4, z4, 0);
#endif
        }
    };
    auto load_b = [&](int k0) -> c16 {
        return *(const c16*)(W + (long)(k0 + b_k) * NDIM + (bn + b_n));
    };
    auto store_b = [&](int buf, c16 w) {         // int8 -> f16, transpose to [n][k]
#pragma unroll
        for (int i = 0; i < 16; ++i)
            Bs[buf][(b_n + i) * BKP + b_k] = (_Float16)(short)w[i];
    };

    // ---- prologue: stage K-tile 0 into buffer 0
    issue_tdm_a(0, 0);
    store_b(0, load_b(0));
    if (wave == 0) __builtin_amdgcn_s_wait_tensorcnt(0);
    __syncthreads();

    int buf = 0;
    for (int k0 = 0; k0 < KDIM; k0 += BK, buf ^= 1) {
        const int  nxt  = buf ^ 1;
        const bool have = (k0 + BK) < KDIM;

        c16 wnext = {};
        if (have) {
            issue_tdm_a(nxt, k0 + BK);           // TDM overlaps the WMMA block
            wnext = load_b(k0 + BK);             // load waits deferred to store_b
        }

        // ---- A fragments (ISA 16-bit A 16x32 layout)
        v16h af[4];
#pragma unroll
        for (int mt = 0; mt < 4; ++mt) {
            const _Float16* p = &As[buf][(wm * 64 + mt * 16 + row16) * AKP + kgrp * 8];
            const v8h lo = *(const v8h*)(p);
            const v8h hi = *(const v8h*)(p + 16);
            v16h a;
#pragma unroll
            for (int i = 0; i < 8; ++i) { a[i] = lo[i]; a[8 + i] = hi[i]; }
            af[mt] = a;
        }
        // ---- B fragments: lanes 0-15 -> K=0..7,16..23, lanes 16-31 -> K=8..15,24..31
        v16h bf[4];
#pragma unroll
        for (int nt = 0; nt < 4; ++nt) {
            const _Float16* p = &Bs[buf][(wn * 64 + nt * 16 + row16) * BKP + kgrp * 8];
            const v8h lo = *(const v8h*)(p);
            const v8h hi = *(const v8h*)(p + 16);
            v16h b;
#pragma unroll
            for (int i = 0; i < 8; ++i) { b[i] = lo[i]; b[8 + i] = hi[i]; }
            bf[nt] = b;
        }

        // ---- 16 WMMAs per K-step
#pragma unroll
        for (int mt = 0; mt < 4; ++mt)
#pragma unroll
            for (int nt = 0; nt < 4; ++nt)
                acc[mt][nt] = __builtin_amdgcn_wmma_f32_16x16x32_f16(
                    false, af[mt], false, bf[nt],
                    (short)0, acc[mt][nt], false, false);

        if (have) {
            store_b(nxt, wnext);                           // cvt + LDS transpose
            if (wave == 0) __builtin_amdgcn_s_wait_tensorcnt(0);  // A tile landed
        }
        __syncthreads();   // one barrier per K-step publishes both buffers
    }

    // ---- epilogue: scale[n] in fp32, emit fp16 per C/D VGPR layout
    const int rsel = kgrp * 8;
#pragma unroll
    for (int nt = 0; nt < 4; ++nt) {
        const long col = bn + wn * 64 + nt * 16 + row16;
        const float sc = (float)S[col];
#pragma unroll
        for (int mt = 0; mt < 4; ++mt) {
            const long rbase = bm + wm * 64 + mt * 16 + rsel;
#pragma unroll
            for (int r = 0; r < 8; ++r)
                O[(rbase + r) * (long)NDIM + col] = (_Float16)(acc[mt][nt][r] * sc);
        }
    }
}

extern "C" void kernel_launch(void* const* d_in, const int* in_sizes, int n_in,
                              void* d_out, int out_size, void* d_ws, size_t ws_size,
                              hipStream_t stream) {
    (void)in_sizes; (void)n_in; (void)out_size; (void)d_ws; (void)ws_size;
    const _Float16*    act   = (const _Float16*)d_in[0];    // [4096,4096] fp16
    const signed char* w     = (const signed char*)d_in[1]; // [4096,14336] int8
    const _Float16*    scale = (const _Float16*)d_in[2];    // [14336] fp16
    _Float16*          out   = (_Float16*)d_out;            // [4096,14336] fp16

    dim3 grid(NDIM / BN, MDIM / BM);   // 112 x 16 blocks
    fused_dq_gemm_f16<<<grid, 256, 0, stream>>>(act, w, scale, out);
}
